// GNN_h_45114336477551
// MI455X (gfx1250) — compile-verified, matrix operand order checked by reading, not run on hardware
//
#include <hip/hip_runtime.h>
#include <hip/hip_bf16.h>

// GNN message passing for MI455X (gfx1250, wave32, WMMA).
//
//   1) zero_f32            : clear m_agg_h / m_agg_w accumulators in d_ws
//   2) pack_wmma_b (x10)   : f32->f16 weights pre-swizzled into the CDNA5
//                            16-bit B-fragment lane layout
//   3) edge_mlp_scatter x2 : 16-edge tiles; WMMA MLPs; layer-2 weights staged
//                            to LDS with global_load_async_to_lds_b128 (wait
//                            hidden behind layer-1 WMMAs); gated atomic scatter
//   4) node_mlp            : 16-node tiles; WMMA MLP -> d_out
//
// Activations use native CDNA5 transcendentals (v_tanh_f32 / v_exp_f32 /
// v_rcp_f32) instead of branchy libm.

#define FEAT     13
#define MSG      64
#define HID      128
#define BATCH    4
#define NNODES   20000
#define EDGES_H  320000
#define EDGES_W  20000
#define WAVES    4
#define EDGE_IN  34     // 2*FEAT + 8
#define EDGE_KP  64     // layer-1 K padded to 2*32
#define NODE_IN  141    // FEAT + 2*MSG
#define NODE_KP  160    // padded to 5*32

typedef __attribute__((ext_vector_type(16))) _Float16 v16h;
typedef __attribute__((ext_vector_type(8)))  _Float16 v8h;
typedef __attribute__((ext_vector_type(8)))  float    v8f;

// ---------------- fast activations on CDNA5 transcendental pipe ------------
static __device__ __forceinline__ float fast_rcp(float x) {
#if __has_builtin(__builtin_amdgcn_rcpf)
  return __builtin_amdgcn_rcpf(x);
#else
  return 1.f / x;
#endif
}

static __device__ __forceinline__ float fast_exp2(float x) {
#if __has_builtin(__builtin_amdgcn_exp2f)
  return __builtin_amdgcn_exp2f(x);
#else
  return exp2f(x);
#endif
}

static __device__ __forceinline__ float fast_tanh(float x) {
#if __has_builtin(__builtin_amdgcn_tanhf)
  return __builtin_amdgcn_tanhf(x);          // v_tanh_f32
#else
  const float e = fast_exp2(x * 2.885390082f);  // exp(2x)
  return 1.f - 2.f * fast_rcp(e + 1.f);
#endif
}

static __device__ __forceinline__ float fast_sigmoid(float x) {
  return fast_rcp(1.f + fast_exp2(x * -1.442695041f));
}

// ---------------- async global->LDS staging (ASYNCcnt) ---------------------
// ROCm 7.2's clang builtin for this op has a toolchain-specific pointer
// signature, so use the ISA mnemonic directly (CDNA5 ISA / 15.18.3):
//   global_load_async_to_lds_b128 vdst(LDS addr), v[addr:addr+1], off
static __device__ __forceinline__ void async_cp_b128(void* lds_dst,
                                                     const void* gsrc) {
#if defined(__gfx1250__)
  asm volatile("global_load_async_to_lds_b128 %0, %1, off"
               :
               : "v"((__attribute__((address_space(3))) void*)lds_dst),
                 "v"(gsrc)
               : "memory");
#else
  *(v8h*)lds_dst = *(const v8h*)gsrc;  // host-pass placeholder
#endif
}

static __device__ __forceinline__ void async_wait0() {
#if __has_builtin(__builtin_amdgcn_s_wait_asynccnt)
  __builtin_amdgcn_s_wait_asynccnt(0);
#elif defined(__gfx1250__)
  asm volatile("s_wait_asynccnt 0x0" ::: "memory");
#endif
}

// ---------------- WMMA helpers ---------------------------------------------
static __device__ __forceinline__ v8f wmma16x16x32(v16h a, v16h b, v8f c) {
  return __builtin_amdgcn_wmma_f32_16x16x32_f16(false, a, false, b, (short)0, c,
                                                false, false);
}

// 16x32 f16 A-fragment from an LDS tile of 16 rows x rowStride halves.
// CDNA5 16-bit A layout: lane L<16 -> row M=L, halves j=0..7 hold K=kb..kb+7,
// j=8..15 hold K=kb+16..kb+23, kb = kc*32 (+8 for lanes 16..31).
static __device__ __forceinline__ v16h load_a_frag(const _Float16* rows,
                                                   int rowStride, int kc,
                                                   int lane) {
  const _Float16* row = rows + (size_t)(lane & 15) * rowStride;
  const int kb = kc * 32 + ((lane & 16) ? 8 : 0);
  v8h lo = *(const v8h*)(row + kb);        // ds_load_b128
  v8h hi = *(const v8h*)(row + kb + 16);   // ds_load_b128
  return __builtin_shufflevector(lo, hi, 0, 1, 2, 3, 4, 5, 6, 7,
                                 8, 9, 10, 11, 12, 13, 14, 15);
}

// Pre-swizzled B fragments: a lane's 16 halves are contiguous (32 bytes).
static __device__ __forceinline__ v16h load_b_frag(const _Float16* packed, int NT,
                                                   int kc, int nt, int lane) {
  return *(const v16h*)(packed + (size_t)(((kc * NT + nt) * 32 + lane) * 16));
}

// ---------------------------------------------------------------------------
// Weight packing: W is (Kreal x Nreal) row-major f32 -> f16 fragments (kc,nt):
// lane l covers n = nt*16 + (l&15); half j maps to
// K = kc*32 + ((l&16)?8:0) + (j<8 ? j : j+8). Zero-pad out-of-range K/N.
// ---------------------------------------------------------------------------
__global__ void pack_wmma_b(const float* __restrict__ src,
                            _Float16* __restrict__ dst,
                            int Kreal, int Nreal, int KC, int NT) {
  const int total = KC * NT * 512;  // halves
  for (int idx = blockIdx.x * blockDim.x + threadIdx.x; idx < total;
       idx += gridDim.x * blockDim.x) {
    const int j    = idx & 15;
    const int lane = (idx >> 4) & 31;
    const int f    = idx >> 9;
    const int nt   = f % NT;
    const int kc   = f / NT;
    const int n    = nt * 16 + (lane & 15);
    const int k    = kc * 32 + ((lane & 16) ? 8 : 0) + (j < 8 ? j : j + 8);
    float v = 0.f;
    if (k < Kreal && n < Nreal) v = src[(size_t)k * Nreal + n];
    dst[idx] = (_Float16)v;
  }
}

__global__ void zero_f32(float* __restrict__ p, long long n4) {
  const long long stride = (long long)gridDim.x * blockDim.x;
  for (long long i = (long long)blockIdx.x * blockDim.x + threadIdx.x; i < n4;
       i += stride)
    ((float4*)p)[i] = make_float4(0.f, 0.f, 0.f, 0.f);
}

// ---------------------------------------------------------------------------
// Edge phase: each wave owns a 16-edge tile.
// ---------------------------------------------------------------------------
__global__ void __launch_bounds__(WAVES * 32)
edge_mlp_scatter(const float* __restrict__ z,
                 const int* __restrict__ eidx, int E,
                 const _Float16* __restrict__ W1m, const float* __restrict__ b1m,
                 const _Float16* __restrict__ W2m, const float* __restrict__ b2m,
                 const _Float16* __restrict__ W1w, const float* __restrict__ b1w,
                 const _Float16* __restrict__ W2w, const float* __restrict__ b2w,
                 float* __restrict__ magg) {
  __shared__ __align__(16) _Float16 sA [WAVES][16 * EDGE_KP];
  __shared__ __align__(16) _Float16 sHm[WAVES][16 * HID];
  __shared__ __align__(16) _Float16 sHw[WAVES][16 * HID];
  __shared__ __align__(16) _Float16 sW2m[8192];  // msg  L2: 4 kc x 4 nt frags
  __shared__ __align__(16) _Float16 sW2w[2048];  // gate L2: 4 kc x 1 nt frags
  __shared__ float sWgt[WAVES][16];
  __shared__ int   sTgt[WAVES][16];

  const int b     = blockIdx.y;
  const int wave  = threadIdx.x >> 5;
  const int lane  = threadIdx.x & 31;
  const int nlo   = lane & 15;
  const int mbase = (lane & 16) ? 8 : 0;
  const int ebase = (blockIdx.x * WAVES + wave) * 16;

  // Kick off async staging of the layer-2 weights; consumed after layer 1,
  // so the ASYNCcnt latency hides behind 32 WMMAs + activations.
  for (int c = threadIdx.x; c < 1024; c += WAVES * 32)
    async_cp_b128(&sW2m[c * 8], W2m + c * 8);
  for (int c = threadIdx.x; c < 256; c += WAVES * 32)
    async_cp_b128(&sW2w[c * 8], W2w + c * 8);

  // layer-1 weights stay in L2; hint them in
  __builtin_prefetch(W1m + lane * 256, 0, 3);
  __builtin_prefetch(W1w + lane * 256, 0, 3);

  // ---- stage A tile: gather z[src], z[tgt], derived edge features ----
  if (lane < 16) {
    const int e = ebase + lane;
    _Float16* row = &sA[wave][lane * EDGE_KP];
    if (e < E) {
      const int src = eidx[(size_t)b * 2 * E + e];
      const int tgt = eidx[(size_t)b * 2 * E + E + e];
      sTgt[wave][lane] = tgt;
      const float* zs = z + ((size_t)b * NNODES + src) * FEAT;
      const float* zt = z + ((size_t)b * NNODES + tgt) * FEAT;
      float f[EDGE_IN];
      #pragma unroll
      for (int i = 0; i < FEAT; ++i) { f[i] = zs[i]; f[FEAT + i] = zt[i]; }
      const float dx = f[0] - f[13], dy = f[1] - f[14], dz = f[2] - f[15];
      const float ax = f[3],  ay = f[4],  az = f[5];
      const float bx = f[16], by = f[17], bz = f[18];
      const float cx = ay * bz - az * by;
      const float cy = az * bx - ax * bz;
      const float cz = ax * by - ay * bx;
      f[26] = dx; f[27] = dy; f[28] = dz;
      f[29] = dx * dx + dy * dy + dz * dz;
      f[30] = cx; f[31] = cy; f[32] = cz;
      f[33] = sqrtf(cx * cx + cy * cy + cz * cz);
      #pragma unroll
      for (int i = 0; i < EDGE_IN; ++i) row[i] = (_Float16)f[i];
      #pragma unroll
      for (int i = EDGE_IN; i < EDGE_KP; ++i) row[i] = (_Float16)0.f;
    } else {
      sTgt[wave][lane] = -1;
      #pragma unroll
      for (int i = 0; i < EDGE_KP; ++i) row[i] = (_Float16)0.f;
    }
  }
  __syncthreads();

  // ---- layer 1: msg + gate MLPs share the (hoisted) A fragments ----
  {
    const v16h a0 = load_a_frag(sA[wave], EDGE_KP, 0, lane);
    const v16h a1 = load_a_frag(sA[wave], EDGE_KP, 1, lane);
    #pragma unroll
    for (int nt = 0; nt < 8; ++nt) {
      v8f accm = {};
      v8f accw = {};
      accm = wmma16x16x32(a0, load_b_frag(W1m, 8, 0, nt, lane), accm);
      accm = wmma16x16x32(a1, load_b_frag(W1m, 8, 1, nt, lane), accm);
      accw = wmma16x16x32(a0, load_b_frag(W1w, 8, 0, nt, lane), accw);
      accw = wmma16x16x32(a1, load_b_frag(W1w, 8, 1, nt, lane), accw);
      const float biasm = b1m[nt * 16 + nlo];
      const float biasw = b1w[nt * 16 + nlo];
      #pragma unroll
      for (int v = 0; v < 8; ++v) {  // D layout: M = v + mbase, N = nlo
        sHm[wave][(size_t)(v + mbase) * HID + nt * 16 + nlo] =
            (_Float16)fast_tanh(accm[v] + biasm);
        sHw[wave][(size_t)(v + mbase) * HID + nt * 16 + nlo] =
            (_Float16)fast_tanh(accw[v] + biasw);
      }
    }
  }
  async_wait0();   // layer-2 weights now resident in LDS
  __syncthreads();

  // ---- gate L2: 128 -> 1 (N padded to 16, only column 0 meaningful) ----
  {
    const v16h a0 = load_a_frag(sHw[wave], HID, 0, lane);
    const v16h a1 = load_a_frag(sHw[wave], HID, 1, lane);
    const v16h a2 = load_a_frag(sHw[wave], HID, 2, lane);
    const v16h a3 = load_a_frag(sHw[wave], HID, 3, lane);
    v8f acc = {};
    acc = wmma16x16x32(a0, load_b_frag(sW2w, 1, 0, 0, lane), acc);
    acc = wmma16x16x32(a1, load_b_frag(sW2w, 1, 1, 0, lane), acc);
    acc = wmma16x16x32(a2, load_b_frag(sW2w, 1, 2, 0, lane), acc);
    acc = wmma16x16x32(a3, load_b_frag(sW2w, 1, 3, 0, lane), acc);
    if (nlo == 0) {  // lanes 0/16 hold column 0 -> rows 0..7 / 8..15
      const float bias = b2w[0];
      #pragma unroll
      for (int v = 0; v < 8; ++v)
        sWgt[wave][v + mbase] = fast_sigmoid(acc[v] + bias);
    }
  }
  __syncthreads();

  float gate[8];
  int   trow[8];
  #pragma unroll
  for (int v = 0; v < 8; ++v) {
    gate[v] = sWgt[wave][v + mbase];
    trow[v] = sTgt[wave][v + mbase];
  }

  // ---- msg L2: 128 -> 64, gated scatter-add into m_agg ----
  {
    const v16h a0 = load_a_frag(sHm[wave], HID, 0, lane);
    const v16h a1 = load_a_frag(sHm[wave], HID, 1, lane);
    const v16h a2 = load_a_frag(sHm[wave], HID, 2, lane);
    const v16h a3 = load_a_frag(sHm[wave], HID, 3, lane);
    #pragma unroll
    for (int nt = 0; nt < 4; ++nt) {
      v8f acc = {};
      acc = wmma16x16x32(a0, load_b_frag(sW2m, 4, 0, nt, lane), acc);
      acc = wmma16x16x32(a1, load_b_frag(sW2m, 4, 1, nt, lane), acc);
      acc = wmma16x16x32(a2, load_b_frag(sW2m, 4, 2, nt, lane), acc);
      acc = wmma16x16x32(a3, load_b_frag(sW2m, 4, 3, nt, lane), acc);
      const float bias = b2m[nt * 16 + nlo];
      #pragma unroll
      for (int v = 0; v < 8; ++v) {
        if (trow[v] >= 0) {
          const float val = (acc[v] + bias) * gate[v];
          unsafeAtomicAdd(
              &magg[((size_t)b * NNODES + trow[v]) * MSG + nt * 16 + nlo], val);
        }
      }
    }
  }
}

// ---------------------------------------------------------------------------
// Node phase: 16-node tiles; A = [z | m_h | m_w] (141 -> pad 160).
// ---------------------------------------------------------------------------
__global__ void __launch_bounds__(WAVES * 32)
node_mlp(const float* __restrict__ z,
         const float* __restrict__ mh, const float* __restrict__ mw,
         const _Float16* __restrict__ W1, const float* __restrict__ b1,
         const _Float16* __restrict__ W2, const float* __restrict__ b2,
         float* __restrict__ out) {
  __shared__ __align__(16) _Float16 sA[WAVES][16 * NODE_KP];
  __shared__ __align__(16) _Float16 sH[WAVES][16 * HID];
  __shared__ __align__(16) _Float16 sW2[2048];  // output layer: 4 kc x 1 nt

  const int b     = blockIdx.y;
  const int wave  = threadIdx.x >> 5;
  const int lane  = threadIdx.x & 31;
  const int nlo   = lane & 15;
  const int mbase = (lane & 16) ? 8 : 0;
  const int nbase = (blockIdx.x * WAVES + wave) * 16;

  for (int c = threadIdx.x; c < 256; c += WAVES * 32)
    async_cp_b128(&sW2[c * 8], W2 + c * 8);
  __builtin_prefetch(W1 + lane * 640, 0, 3);

  if (lane < 16) {
    const int node = nbase + lane;
    _Float16* row = &sA[wave][lane * NODE_KP];
    if (node < NNODES) {
      const float* zp  = z  + ((size_t)b * NNODES + node) * FEAT;
      const float* mhp = mh + ((size_t)b * NNODES + node) * MSG;
      const float* mwp = mw + ((size_t)b * NNODES + node) * MSG;
      #pragma unroll
      for (int i = 0; i < FEAT; ++i) row[i] = (_Float16)zp[i];
      #pragma unroll
      for (int i = 0; i < MSG; ++i) row[FEAT + i] = (_Float16)mhp[i];
      #pragma unroll
      for (int i = 0; i < MSG; ++i) row[FEAT + MSG + i] = (_Float16)mwp[i];
      #pragma unroll
      for (int i = NODE_IN; i < NODE_KP; ++i) row[i] = (_Float16)0.f;
    } else {
      #pragma unroll
      for (int i = 0; i < NODE_KP; ++i) row[i] = (_Float16)0.f;
    }
  }
  __syncthreads();

  {
    v16h a[5];
    #pragma unroll
    for (int kc = 0; kc < 5; ++kc)
      a[kc] = load_a_frag(sA[wave], NODE_KP, kc, lane);
    #pragma unroll
    for (int nt = 0; nt < 8; ++nt) {
      v8f acc = {};
      #pragma unroll
      for (int kc = 0; kc < 5; ++kc)
        acc = wmma16x16x32(a[kc], load_b_frag(W1, 8, kc, nt, lane), acc);
      const float bias = b1[nt * 16 + nlo];
      #pragma unroll
      for (int v = 0; v < 8; ++v)
        sH[wave][(size_t)(v + mbase) * HID + nt * 16 + nlo] =
            (_Float16)fast_tanh(acc[v] + bias);
    }
  }
  async_wait0();
  __syncthreads();

  {
    const v16h a0 = load_a_frag(sH[wave], HID, 0, lane);
    const v16h a1 = load_a_frag(sH[wave], HID, 1, lane);
    const v16h a2 = load_a_frag(sH[wave], HID, 2, lane);
    const v16h a3 = load_a_frag(sH[wave], HID, 3, lane);
    v8f acc = {};
    acc = wmma16x16x32(a0, load_b_frag(sW2, 1, 0, 0, lane), acc);
    acc = wmma16x16x32(a1, load_b_frag(sW2, 1, 1, 0, lane), acc);
    acc = wmma16x16x32(a2, load_b_frag(sW2, 1, 2, 0, lane), acc);
    acc = wmma16x16x32(a3, load_b_frag(sW2, 1, 3, 0, lane), acc);
    if (nlo < FEAT) {
      const float bias = b2[nlo];
      #pragma unroll
      for (int v = 0; v < 8; ++v) {
        const int node = nbase + v + mbase;
        if (node < NNODES)
          out[((size_t)b * NNODES + node) * FEAT + nlo] = acc[v] + bias;
      }
    }
  }
}

// ---------------------------------------------------------------------------
// Host glue. d_in order (top-level dict order, params leaves key-sorted):
//  0 z_h  1 edge_h  2 edge_w
//  3..6   edge.{W1,W2,b1,b2}       7..10  edge_world.{W1,W2,b1,b2}
// 11..14  node.{W1,W2,b1,b2}      15..18  weight.{W1,W2,b1,b2}
// 19..22  weight_world.{W1,W2,b1,b2}
// ---------------------------------------------------------------------------
extern "C" void kernel_launch(void* const* d_in, const int* in_sizes, int n_in,
                              void* d_out, int out_size, void* d_ws, size_t ws_size,
                              hipStream_t stream) {
  (void)in_sizes; (void)n_in; (void)out_size; (void)ws_size;

  const float* z    = (const float*)d_in[0];
  const int*   eh   = (const int*)d_in[1];
  const int*   ew   = (const int*)d_in[2];
  const float* eW1  = (const float*)d_in[3];
  const float* eW2  = (const float*)d_in[4];
  const float* eb1  = (const float*)d_in[5];
  const float* eb2  = (const float*)d_in[6];
  const float* ewW1 = (const float*)d_in[7];
  const float* ewW2 = (const float*)d_in[8];
  const float* ewb1 = (const float*)d_in[9];
  const float* ewb2 = (const float*)d_in[10];
  const float* nW1  = (const float*)d_in[11];
  const float* nW2  = (const float*)d_in[12];
  const float* nb1  = (const float*)d_in[13];
  const float* nb2  = (const float*)d_in[14];
  const float* wW1  = (const float*)d_in[15];
  const float* wW2  = (const float*)d_in[16];
  const float* wb1  = (const float*)d_in[17];
  const float* wb2  = (const float*)d_in[18];
  const float* wwW1 = (const float*)d_in[19];
  const float* wwW2 = (const float*)d_in[20];
  const float* wwb1 = (const float*)d_in[21];
  const float* wwb2 = (const float*)d_in[22];

  // workspace layout: [m_agg_h f32][m_agg_w f32][packed f16 weights]
  const size_t MAGG = (size_t)BATCH * NNODES * MSG;   // 5.12M floats each
  float* mh = (float*)d_ws;
  float* mw = mh + MAGG;
  _Float16* pk    = (_Float16*)(mw + MAGG);
  _Float16* pW1e  = pk;              // 2*8*512 = 8192 halves
  _Float16* pW1w  = pW1e + 8192;
  _Float16* pW1ew = pW1w + 8192;
  _Float16* pW1ww = pW1ew + 8192;
  _Float16* pW2e  = pW1ww + 8192;    // 4*4*512 = 8192
  _Float16* pW2ew = pW2e + 8192;
  _Float16* pW2w  = pW2ew + 8192;    // 4*1*512 = 2048
  _Float16* pW2ww = pW2w + 2048;
  _Float16* pWn1  = pW2ww + 2048;    // 5*8*512 = 20480
  _Float16* pWn2  = pWn1 + 20480;    // 4*1*512 = 2048

  zero_f32<<<2048, 256, 0, stream>>>(mh, (long long)(2 * MAGG / 4));

  auto packLaunch = [&](const float* src, _Float16* dst, int Kr, int Nr, int KC,
                        int NT) {
    const int total = KC * NT * 512;
    pack_wmma_b<<<(total + 255) / 256, 256, 0, stream>>>(src, dst, Kr, Nr, KC, NT);
  };
  packLaunch(eW1,  pW1e,  EDGE_IN, HID, 2, 8);
  packLaunch(wW1,  pW1w,  EDGE_IN, HID, 2, 8);
  packLaunch(ewW1, pW1ew, EDGE_IN, HID, 2, 8);
  packLaunch(wwW1, pW1ww, EDGE_IN, HID, 2, 8);
  packLaunch(eW2,  pW2e,  HID, MSG, 4, 4);
  packLaunch(ewW2, pW2ew, HID, MSG, 4, 4);
  packLaunch(wW2,  pW2w,  HID, 1,   4, 1);
  packLaunch(wwW2, pW2ww, HID, 1,   4, 1);
  packLaunch(nW1,  pWn1,  NODE_IN, HID, 5, 8);
  packLaunch(nW2,  pWn2,  HID, FEAT,   4, 1);

  const dim3 blk(WAVES * 32);
  const int TILE = WAVES * 16;
  dim3 gE((EDGES_H + TILE - 1) / TILE, BATCH);
  edge_mlp_scatter<<<gE, blk, 0, stream>>>(z, eh, EDGES_H, pW1e, eb1, pW2e, eb2,
                                           pW1w, wb1, pW2w, wb2, mh);
  dim3 gW((EDGES_W + TILE - 1) / TILE, BATCH);
  edge_mlp_scatter<<<gW, blk, 0, stream>>>(z, ew, EDGES_W, pW1ew, ewb1, pW2ew, ewb2,
                                           pW1ww, wwb1, pW2ww, wwb2, mw);
  dim3 gN((NNODES + TILE - 1) / TILE, BATCH);
  node_mlp<<<gN, blk, 0, stream>>>(z, mh, mw, pWn1, nb1, pWn2, nb2, (float*)d_out);
}
